// Head_31207232373254
// MI455X (gfx1250) — matrix operand order repaired and checked by reading, not compiled, not run on hardware
//
#include <hip/hip_runtime.h>
#include <hip/hip_bf16.h>

// Problem constants (from reference): B=4, T=4096, C=1024, H=64
#define BB 4
#define TT 4096
#define CC 1024
#define HH 64

typedef __bf16 bf16_t;
typedef __attribute__((ext_vector_type(8)))  __bf16 v8bf;
typedef __attribute__((ext_vector_type(16))) __bf16 v16bf;
typedef __attribute__((ext_vector_type(8)))  float  v8f;

static __device__ __forceinline__ v16bf join16(v8bf lo, v8bf hi) {
    union { v8bf p[2]; v16bf v; } u;
    u.p[0] = lo; u.p[1] = hi;
    return u.v;
}
static __device__ __forceinline__ v8bf ld8(const bf16_t* p) {
    return *(const v8bf*)p;   // 16B contiguous, caller guarantees alignment
}
static __device__ __forceinline__ v8bf cvt8(const float* p) {
    v8bf r;
#pragma unroll
    for (int i = 0; i < 8; ++i) r[i] = (bf16_t)p[i];
    return r;
}
static __device__ __forceinline__ v8f zero8() {
    v8f z = {0.f, 0.f, 0.f, 0.f, 0.f, 0.f, 0.f, 0.f};
    return z;
}
static __device__ __forceinline__ v8f wmma_bf16(v16bf a, v16bf b, v8f c) {
    // D(f32 16x16) = A(bf16 16x32) * B(bf16 32x16) + C
    return __builtin_amdgcn_wmma_f32_16x16x32_bf16(false, a, false, b, (short)0, c, false, false);
}

// ---------------------------------------------------------------------------
// Kernel 0: convert the three weight matrices [C][H] f32 -> bf16 transposed
// to [H][C] (WMMA B-fragment: lane=N column, contiguous K).
// grid.y selects the matrix -> no div/mod.
// ---------------------------------------------------------------------------
__global__ void wcvt_kernel(const float* __restrict__ wq,
                            const float* __restrict__ wk,
                            const float* __restrict__ wv,
                            bf16_t* __restrict__ wb) {
    const int m = blockIdx.y;
    const int j = blockIdx.x * blockDim.x + threadIdx.x;   // 0 .. C*H-1
    const float* src = (m == 0) ? wq : (m == 1) ? wk : wv;
    wb[(size_t)m * CC * HH + (size_t)(j & (HH - 1)) * CC + (j >> 6)] = (bf16_t)src[j];
}

// ---------------------------------------------------------------------------
// Kernel 1: fused projections Q,K,V = x @ {Wq,Wk,Wv} via WMMA bf16.
//   grid: (B*T)/64 workgroups of 128 threads (4 waves); wave -> 16 token rows.
//   One pass over x: per 32-wide K-step, one A fragment feeds 12 WMMAs
//   (3 matrices x 4 h-tiles). Store epilogues are branch-free.
//   Q: row-major bf16 with sqrt(C)=32 folded in.  K: row-major [B][T][H].
//   V: transposed [B][H][T].
// ---------------------------------------------------------------------------
__global__ __launch_bounds__(128) void proj_kernel(const float* __restrict__ x,
                                                   const bf16_t* __restrict__ wb,
                                                   bf16_t* __restrict__ qb,
                                                   bf16_t* __restrict__ ktp,
                                                   bf16_t* __restrict__ vb) {
    const int lane  = threadIdx.x & 31;
    const int wave  = threadIdx.x >> 5;
    const int hi    = lane >> 4;
    const int ln    = lane & 15;
    const int rbase = blockIdx.x * 64 + wave * 16;
    const int arow  = rbase + ln;      // A layout: both half-waves map M=ln
    const int kboff = hi ? 8 : 0;      // 16-bit A layout K-chunk offset
    const float* xr = x + (size_t)arow * CC;

    v8f accq[4] = {zero8(), zero8(), zero8(), zero8()};
    v8f acck[4] = {zero8(), zero8(), zero8(), zero8()};
    v8f accv[4] = {zero8(), zero8(), zero8(), zero8()};

    for (int kc = 0; kc < CC; kc += 32) {
        // A fragment: x[row][kc+kboff .. +7] and [kc+kboff+16 .. +7], f32->bf16
        const v16bf afrag =
            join16(cvt8(xr + kc + kboff), cvt8(xr + kc + kboff + 16));

        // B fragments: lane = N column (ln), K chunks kboff and kboff+16
        const bf16_t* wrq = wb + (size_t)ln * CC + kc + kboff;
        const bf16_t* wrk = wrq + (size_t)CC * HH;
        const bf16_t* wrv = wrq + (size_t)2 * CC * HH;

        {   // ---- Wq group: 4 load-pairs then 4 WMMAs (scheduler overlaps)
            v16bf f0 = join16(ld8(wrq +  0 * CC), ld8(wrq +  0 * CC + 16));
            v16bf f1 = join16(ld8(wrq + 16 * CC), ld8(wrq + 16 * CC + 16));
            v16bf f2 = join16(ld8(wrq + 32 * CC), ld8(wrq + 32 * CC + 16));
            v16bf f3 = join16(ld8(wrq + 48 * CC), ld8(wrq + 48 * CC + 16));
            accq[0] = wmma_bf16(afrag, f0, accq[0]);
            accq[1] = wmma_bf16(afrag, f1, accq[1]);
            accq[2] = wmma_bf16(afrag, f2, accq[2]);
            accq[3] = wmma_bf16(afrag, f3, accq[3]);
        }
        {   // ---- Wk group
            v16bf f0 = join16(ld8(wrk +  0 * CC), ld8(wrk +  0 * CC + 16));
            v16bf f1 = join16(ld8(wrk + 16 * CC), ld8(wrk + 16 * CC + 16));
            v16bf f2 = join16(ld8(wrk + 32 * CC), ld8(wrk + 32 * CC + 16));
            v16bf f3 = join16(ld8(wrk + 48 * CC), ld8(wrk + 48 * CC + 16));
            acck[0] = wmma_bf16(afrag, f0, acck[0]);
            acck[1] = wmma_bf16(afrag, f1, acck[1]);
            acck[2] = wmma_bf16(afrag, f2, acck[2]);
            acck[3] = wmma_bf16(afrag, f3, acck[3]);
        }
        {   // ---- Wv group
            v16bf f0 = join16(ld8(wrv +  0 * CC), ld8(wrv +  0 * CC + 16));
            v16bf f1 = join16(ld8(wrv + 16 * CC), ld8(wrv + 16 * CC + 16));
            v16bf f2 = join16(ld8(wrv + 32 * CC), ld8(wrv + 32 * CC + 16));
            v16bf f3 = join16(ld8(wrv + 48 * CC), ld8(wrv + 48 * CC + 16));
            accv[0] = wmma_bf16(afrag, f0, accv[0]);
            accv[1] = wmma_bf16(afrag, f1, accv[1]);
            accv[2] = wmma_bf16(afrag, f2, accv[2]);
            accv[3] = wmma_bf16(afrag, f3, accv[3]);
        }
    }

    // C/D layout: VGPR r -> row (r + 8*hi), lane -> column (ln).
    // Three branch-free store loops.
#pragma unroll
    for (int r = 0; r < 8; ++r) {
        const int tok = rbase + r + hi * 8;
        bf16_t* qrow = qb + (size_t)tok * HH + ln;
#pragma unroll
        for (int ht = 0; ht < 4; ++ht)
            qrow[ht * 16] = (bf16_t)(accq[ht][r] * 32.0f);   // fold sqrt(C)
    }
#pragma unroll
    for (int r = 0; r < 8; ++r) {
        const int tok  = rbase + r + hi * 8;
        const int bidx = tok >> 12;            // /T
        const int t    = tok & (TT - 1);
        bf16_t* kcol = ktp + (size_t)tok * HH + ln;
#pragma unroll
        for (int ht = 0; ht < 4; ++ht)
            kcol[ht * 16] = (bf16_t)acck[ht][r];
    }
#pragma unroll
    for (int r = 0; r < 8; ++r) {
        const int tok = rbase + r + hi * 8;
        bf16_t* vrow = vb + ((size_t)((tok >> 12) * HH + ln) << 12) + (tok & (TT - 1));
#pragma unroll
        for (int ht = 0; ht < 4; ++ht)
            vrow[(size_t)(ht * 16) << 12] = (bf16_t)accv[ht][r];
    }
}

// ---------------------------------------------------------------------------
// Kernel 2: causal flash attention.
//   grid: B*(T/64) workgroups of 128 threads; wave -> 16 query rows.
//   Per 32-key tile: S = Q K^T (4 WMMAs), online softmax (shfl reductions in
//   16-lane groups matching the C/D row layout), P -> bf16 via per-wave LDS
//   re-layout, O += P V (4 WMMAs). Next tile's K/V prefetched while softmax
//   VALU work runs. Trip count is workgroup-uniform so __syncthreads is legal.
// ---------------------------------------------------------------------------
__global__ __launch_bounds__(128) void attn_kernel(const bf16_t* __restrict__ qb,
                                                   const bf16_t* __restrict__ ktp,
                                                   const bf16_t* __restrict__ vb,
                                                   float* __restrict__ out) {
    __shared__ bf16_t pbuf[4][16][32];   // per-wave P tile (bf16, row-major)

    const int lane  = threadIdx.x & 31;
    const int wave  = threadIdx.x >> 5;
    const int hi    = lane >> 4;
    const int ln    = lane & 15;
    const int kboff = hi ? 8 : 0;
    const int b     = blockIdx.x >> 6;   // T/64 = 64 row-tiles per batch
    const int qt    = blockIdx.x & 63;
    const int qwb   = qt * 64 + wave * 16;

    const bf16_t* qp = qb  + (size_t)b * TT * HH;
    const bf16_t* kp = ktp + (size_t)b * HH * TT;
    const bf16_t* vp = vb  + (size_t)b * TT * HH;
    float*        op = out + (size_t)b * TT * HH;

    // Q fragments (A layout), covering h = 0..31 and 32..63
    const bf16_t* qrp = qp + (size_t)(qwb + ln) * HH;
    const v16bf qf0 = join16(ld8(qrp + kboff),      ld8(qrp + kboff + 16));
    const v16bf qf1 = join16(ld8(qrp + 32 + kboff), ld8(qrp + 32 + kboff + 16));

    float mrow[8], lrow[8];
    v8f o[4] = {zero8(), zero8(), zero8(), zero8()};
#pragma unroll
    for (int r = 0; r < 8; ++r) { mrow[r] = -3.0e38f; lrow[r] = 0.0f; }

    const bf16_t* kr0 = kp + (size_t)ln * HH + kboff;    // key = ln, h chunk kboff
    const bf16_t* kr1 = kr0 + 16 * HH;                   // key = 16+ln

    const int ntiles = qt * 2 + 2;   // ceil((qt*64 + 64)/32), block-uniform
    for (int ktile = 0; ktile < ntiles; ++ktile) {
        const int ktb = ktile * 32;

        // ---- S = Q K^T : two 16x16 f32 tiles (keys ktb..+15 and +16..+31)
        v8f s0 = zero8(), s1 = zero8();
        {
            v16bf kf;
            kf = join16(ld8(kr0 + ktb * HH),      ld8(kr0 + ktb * HH + 16));
            s0 = wmma_bf16(qf0, kf, s0);
            kf = join16(ld8(kr1 + ktb * HH),      ld8(kr1 + ktb * HH + 16));
            s1 = wmma_bf16(qf0, kf, s1);
            kf = join16(ld8(kr0 + ktb * HH + 32), ld8(kr0 + ktb * HH + 48));
            s0 = wmma_bf16(qf1, kf, s0);
            kf = join16(ld8(kr1 + ktb * HH + 32), ld8(kr1 + ktb * HH + 48));
            s1 = wmma_bf16(qf1, kf, s1);
        }

        // ---- prefetch next tile's K/V while softmax VALU runs
        if (ktile + 1 < ntiles) {
            const int nktb = ktb + 32;
            __builtin_prefetch(kr0 + nktb, 0, 1);
            __builtin_prefetch(kr1 + nktb, 0, 1);
            __builtin_prefetch(vp + (size_t)(nktb + lane) * HH, 0, 1);
        }

        // ---- causal mask + online softmax (per D-layout row)
        const int n0 = ktb + ln;
        const int n1 = n0 + 16;
#pragma unroll
        for (int r = 0; r < 8; ++r) {
            const int qrow = qwb + r + hi * 8;
            float v0 = (n0 <= qrow) ? s0[r] : -1.0e30f;
            float v1 = (n1 <= qrow) ? s1[r] : -1.0e30f;

            float mx = fmaxf(v0, v1);
#pragma unroll
            for (int msk = 1; msk < 16; msk <<= 1)
                mx = fmaxf(mx, __shfl_xor(mx, msk, 32));

            const float mnew  = fmaxf(mrow[r], mx);
            const float alpha = __expf(mrow[r] - mnew);
            mrow[r] = mnew;

            const float p0 = __expf(v0 - mnew);
            const float p1 = __expf(v1 - mnew);
            float rs = p0 + p1;
#pragma unroll
            for (int msk = 1; msk < 16; msk <<= 1)
                rs += __shfl_xor(rs, msk, 32);
            lrow[r] = lrow[r] * alpha + rs;

            const int mr = r + hi * 8;
            pbuf[wave][mr][ln]      = (bf16_t)p0;
            pbuf[wave][mr][ln + 16] = (bf16_t)p1;

#pragma unroll
            for (int ht = 0; ht < 4; ++ht) o[ht][r] *= alpha;
        }
        __syncthreads();

        // ---- O += P V : P as A fragment from LDS (row = ln, K = key index)
        const bf16_t* prp = &pbuf[wave][ln][0];
        const v16bf pf = join16(ld8(prp + kboff), ld8(prp + kboff + 16));
        const bf16_t* vr = vp + (size_t)ln * TT + ktb + kboff;   // lane = h column
#pragma unroll
        for (int ht = 0; ht < 4; ++ht) {
            v16bf vf = join16(ld8(vr + (size_t)(ht * 16) * TT), ld8(vr + (size_t)(ht * 16) * TT + 16));
            o[ht] = wmma_bf16(pf, vf, o[ht]);
        }
        __syncthreads();
    }

    // ---- epilogue: divide by row sums, store f32 [B][T][H]
#pragma unroll
    for (int r = 0; r < 8; ++r) {
        const int qrow = qwb + r + hi * 8;
        const float inv = 1.0f / lrow[r];
#pragma unroll
        for (int ht = 0; ht < 4; ++ht)
            op[(size_t)qrow * HH + ht * 16 + ln] = o[ht][r] * inv;
    }
}

// ---------------------------------------------------------------------------
extern "C" void kernel_launch(void* const* d_in, const int* in_sizes, int n_in,
                              void* d_out, int out_size, void* d_ws, size_t ws_size,
                              hipStream_t stream) {
    const float* x  = (const float*)d_in[0];
    const float* wq = (const float*)d_in[1];
    const float* wk = (const float*)d_in[2];
    const float* wv = (const float*)d_in[3];

    // Workspace layout (bf16): Wb[3*C*H] | Q[B*T*H] | Kt[B*H*T] | V[B*T*H]
    bf16_t* wb = (bf16_t*)d_ws;
    bf16_t* qb = wb + (size_t)3 * CC * HH;
    bf16_t* kt = qb + (size_t)BB * TT * HH;
    bf16_t* vb = kt + (size_t)BB * TT * HH;

    dim3 wgrid((CC * HH) / 256, 3, 1);
    wcvt_kernel<<<wgrid, 256, 0, stream>>>(wq, wk, wv, wb);
    proj_kernel<<<(BB * TT) / 64, 128, 0, stream>>>(x, wb, qb, kt, vb);
    attn_kernel<<<(BB * TT) / 64, 128, 0, stream>>>(qb, kt, vb, (float*)d_out);
}
